// SP_CAM_Model_16140487098909
// MI455X (gfx1250) — matrix operand order repaired and checked by reading, not compiled, not run on hardware
//
#include <hip/hip_runtime.h>

typedef __attribute__((ext_vector_type(16))) _Float16 v16h;
typedef __attribute__((ext_vector_type(8)))  _Float16 v8h;
typedef __attribute__((ext_vector_type(8)))  float    v8f;

// ---------------- workspace layout (bytes, all 256-aligned) ----------------
static const size_t OFF_Q1   = 0;           // fp32 [8][32][128][128]   16,777,216
static const size_t OFF_Q2   = 16777216;    // fp32 [8][64][32][32]      2,097,152
static const size_t OFF_Q3   = 18874368;    // fp32 [8][128][32][32]     4,194,304
static const size_t OFF_Q4   = 23068672;    // fp32 [8][128][32][32]     4,194,304
static const size_t OFF_FEAT = 27262976;    // f16  [8][34][34][2176]   40,247,296 (zero-padded NHWC)
static const size_t OFF_AW1  = 67510272;    // f16  [9][1024][2176]     40,108,032
static const size_t OFF_A2   = 107618304;   // f16  [256][1024]            524,288
static const size_t OFF_D1   = 108142592;   // f16  [8192][1024]        16,777,216
static const size_t OFF_D2   = 124919808;   // f16  [8192][256]          4,194,304
static const size_t OFF_D3   = 129114112;   // fp32 [8192][18]             589,824
static const size_t OFF_LOG  = 129703936;   // fp32 [8][21][32][32]        688,128
static const size_t OFF_GAP  = 130392064;   // fp32 [8][2048]               65,536
// total ~130.5 MB

// ---------------- WMMA fragment helpers ----------------
// A (16x32 f16): lane (m = lane&15, khalf = lane>>4) holds K = khalf*16+{0..15}
// -> one contiguous 32B load from row-major [M][K].
__device__ __forceinline__ v16h load_a_frag(const _Float16* __restrict__ p, int lhalf) {
  return *(const v16h*)(p + lhalf * 16);
}
// B (32x16 f16) from Bt[n][k] layout: lane (n = lane&15, khalf) holds
// k = khalf*16+{0..15} -> one contiguous 32B load.
__device__ __forceinline__ v16h load_b_frag(const _Float16* __restrict__ row, int kc, int lhalf) {
  return *(const v16h*)(row + kc + lhalf * 16);
}
// C/D (16x16 f32): vgpr r, lane -> m = mtile + (lane>>4)*8 + r, n = ntile + (lane&15)
__device__ __forceinline__ void store_tile_relu_f16(
    _Float16* __restrict__ out, int Mstride, v8f acc,
    int n, int mbase, const float* __restrict__ bias) {
  v8h h;
  #pragma unroll
  for (int r = 0; r < 8; ++r) {
    float v = acc[r] + bias[mbase + r];
    h[r] = (_Float16)(v > 0.f ? v : 0.f);
  }
  *(v8h*)(out + (size_t)n * Mstride + mbase) = h;
}

// ---------------- big 3x3 conv as implicit GEMM on WMMA ----------------
// D[m,n] = relu( sum_{r,k} Aw[r][m][k] * feat[b][y+ky][x+kx][k] + bias[m] )
// M=1024, N=8192 (b,y,x), K=9*2176.  Block: 128(M)x64(N), wave: 32x32 (2x2 tiles).
__global__ __launch_bounds__(256)
void conv3x3_wmma_kernel(const _Float16* __restrict__ Aw,
                         const _Float16* __restrict__ feat,
                         const float* __restrict__ bias,
                         _Float16* __restrict__ out) {
  const int lane  = threadIdx.x & 31;
  const int wv    = threadIdx.x >> 5;
  const int wm    = wv & 3;            // 4 waves over M
  const int wn    = wv >> 2;           // 2 waves over N
  const int lidx  = lane & 15;
  const int lhalf = lane >> 4;
  const int bm = (blockIdx.x & 7) * 128;
  const int bn = (blockIdx.x >> 3) * 64;
  const int m0 = bm + wm * 32;
  const int n0 = bn + wn * 32;

  const int nA = n0 + lidx, nB = nA + 16;
  const _Float16* brow0 = feat +
      ((size_t)(((nA >> 10) * 34 + ((nA >> 5) & 31)) * 34) + (nA & 31)) * 2176;
  const _Float16* brow1 = feat +
      ((size_t)(((nB >> 10) * 34 + ((nB >> 5) & 31)) * 34) + (nB & 31)) * 2176;
  const _Float16* arow = Aw + (size_t)(m0 + lidx) * 2176;

  v8f c00 = {}, c01 = {}, c10 = {}, c11 = {};
  for (int r = 0; r < 9; ++r) {
    const int ky = r / 3, kx = r % 3;
    const _Float16* pa0 = arow + (size_t)r * (1024 * 2176);
    const _Float16* pa1 = pa0 + (size_t)16 * 2176;
    const _Float16* pb0 = brow0 + (size_t)(ky * 34 + kx) * 2176;
    const _Float16* pb1 = brow1 + (size_t)(ky * 34 + kx) * 2176;
    for (int kc = 0; kc < 2176; kc += 32) {
      v16h a0 = load_a_frag(pa0 + kc, lhalf);
      v16h a1 = load_a_frag(pa1 + kc, lhalf);
      v16h b0 = load_b_frag(pb0, kc, lhalf);
      v16h b1 = load_b_frag(pb1, kc, lhalf);
      c00 = __builtin_amdgcn_wmma_f32_16x16x32_f16(false, a0, false, b0, (short)0, c00, false, false);
      c01 = __builtin_amdgcn_wmma_f32_16x16x32_f16(false, a0, false, b1, (short)0, c01, false, false);
      c10 = __builtin_amdgcn_wmma_f32_16x16x32_f16(false, a1, false, b0, (short)0, c10, false, false);
      c11 = __builtin_amdgcn_wmma_f32_16x16x32_f16(false, a1, false, b1, (short)0, c11, false, false);
    }
  }
  const int mb0 = m0 + lhalf * 8, mb1 = m0 + 16 + lhalf * 8;
  store_tile_relu_f16(out, 1024, c00, n0 + lidx,      mb0, bias);
  store_tile_relu_f16(out, 1024, c01, n0 + 16 + lidx, mb0, bias);
  store_tile_relu_f16(out, 1024, c10, n0 + lidx,      mb1, bias);
  store_tile_relu_f16(out, 1024, c11, n0 + 16 + lidx, mb1, bias);
}

// ---------------- 1x1 conv (d2): plain WMMA GEMM, A[M][K] x Bt[N][K] ----------------
__global__ __launch_bounds__(256)
void gemm_nk_wmma_kernel(const _Float16* __restrict__ A,
                         const _Float16* __restrict__ Bt,
                         const float* __restrict__ bias,
                         _Float16* __restrict__ out,
                         int M, int K, int mblocks) {
  const int lane  = threadIdx.x & 31;
  const int wv    = threadIdx.x >> 5;
  const int wm    = wv & 3;
  const int wn    = wv >> 2;
  const int lidx  = lane & 15;
  const int lhalf = lane >> 4;
  const int bm = (blockIdx.x % mblocks) * 128;
  const int bn = (blockIdx.x / mblocks) * 64;
  const int m0 = bm + wm * 32;
  const int n0 = bn + wn * 32;

  const _Float16* pa0 = A + (size_t)(m0 + lidx) * K;
  const _Float16* pa1 = pa0 + (size_t)16 * K;
  const _Float16* pb0 = Bt + (size_t)(n0 + lidx) * K;
  const _Float16* pb1 = pb0 + (size_t)16 * K;

  v8f c00 = {}, c01 = {}, c10 = {}, c11 = {};
  for (int kc = 0; kc < K; kc += 32) {
    v16h a0 = load_a_frag(pa0 + kc, lhalf);
    v16h a1 = load_a_frag(pa1 + kc, lhalf);
    v16h b0 = load_b_frag(pb0, kc, lhalf);
    v16h b1 = load_b_frag(pb1, kc, lhalf);
    c00 = __builtin_amdgcn_wmma_f32_16x16x32_f16(false, a0, false, b0, (short)0, c00, false, false);
    c01 = __builtin_amdgcn_wmma_f32_16x16x32_f16(false, a0, false, b1, (short)0, c01, false, false);
    c10 = __builtin_amdgcn_wmma_f32_16x16x32_f16(false, a1, false, b0, (short)0, c10, false, false);
    c11 = __builtin_amdgcn_wmma_f32_16x16x32_f16(false, a1, false, b1, (short)0, c11, false, false);
  }
  const int mb0 = m0 + lhalf * 8, mb1 = m0 + 16 + lhalf * 8;
  store_tile_relu_f16(out, M, c00, n0 + lidx,      mb0, bias);
  store_tile_relu_f16(out, M, c01, n0 + 16 + lidx, mb0, bias);
  store_tile_relu_f16(out, M, c10, n0 + lidx,      mb1, bias);
  store_tile_relu_f16(out, M, c11, n0 + 16 + lidx, mb1, bias);
}

// ---------------- naive fp32 conv+relu (q1..q4; small FLOPs, L2-resident) ----------------
__global__ void conv_naive_relu_kernel(const float* __restrict__ in,
                                       const float* __restrict__ w,
                                       const float* __restrict__ bias,
                                       float* __restrict__ out,
                                       int B, int Cin, int Hin, int Cout, int Hout,
                                       int kk, int stride, int pad) {
  int idx = blockIdx.x * 256 + threadIdx.x;
  int total = B * Cout * Hout * Hout;
  if (idx >= total) return;
  int x = idx % Hout; int t = idx / Hout;
  int y = t % Hout;   t /= Hout;
  int oc = t % Cout;  int b = t / Cout;
  float acc = bias[oc];
  for (int c = 0; c < Cin; ++c) {
    const float* ip = in + ((size_t)(b * Cin + c) * Hin) * Hin;
    const float* wp = w + ((size_t)(oc * Cin + c) * kk) * kk;
    for (int ky = 0; ky < kk; ++ky) {
      int iy = y * stride + ky - pad;
      if ((unsigned)iy >= (unsigned)Hin) continue;
      for (int kx = 0; kx < kk; ++kx) {
        int ix = x * stride + kx - pad;
        if ((unsigned)ix >= (unsigned)Hin) continue;
        acc += ip[(size_t)iy * Hin + ix] * wp[ky * kk + kx];
      }
    }
  }
  out[idx] = acc > 0.f ? acc : 0.f;
}

// ---------------- packing / layout kernels ----------------
__global__ void zero_h_kernel(_Float16* __restrict__ p, int n8) {
  int i = blockIdx.x * 256 + threadIdx.x;
  if (i >= n8) return;
  v8h z;
  #pragma unroll
  for (int r = 0; r < 8; ++r) z[r] = (_Float16)0.f;
  *(v8h*)(p + (size_t)i * 8) = z;
}

// NCHW fp32 (x5 ++ q4) -> padded NHWC f16 feat, via LDS 32x32 transpose tiles
__global__ __launch_bounds__(256)
void pack_feat_kernel(const float* __restrict__ x5, const float* __restrict__ q4,
                      _Float16* __restrict__ feat) {
  __shared__ float tile[32][33];
  int blk = blockIdx.x;
  int pt = blk & 31; blk >>= 5;
  int ct = blk % 68; int b = blk / 68;
  int c0 = ct * 32, p0 = pt * 32;
  for (int t = threadIdx.x; t < 1024; t += 256) {
    int ci = t >> 5, pi = t & 31;
    int c = c0 + ci;
    float v = (c < 2048)
        ? x5[((size_t)(b * 2048 + c) << 10) + p0 + pi]
        : q4[((size_t)(b * 128 + (c - 2048)) << 10) + p0 + pi];
    tile[ci][pi] = v;
  }
  __syncthreads();
  for (int t = threadIdx.x; t < 1024; t += 256) {
    int pi = t >> 5, ci = t & 31;
    int p = p0 + pi, y = p >> 5, x = p & 31;
    feat[((size_t)((b * 34 + y + 1) * 34) + (x + 1)) * 2176 + c0 + ci] =
        (_Float16)tile[ci][pi];
  }
}

// tw1 OIHW fp32 -> Aw f16 [9][1024][2176]
__global__ void pack_w1_kernel(const float* __restrict__ tw1, _Float16* __restrict__ Aw) {
  int idx = blockIdx.x * 256 + threadIdx.x;
  if (idx >= 1024 * 2176) return;
  int c = idx % 2176, oc = idx / 2176;
  const float* src = tw1 + (size_t)idx * 9;
  for (int r = 0; r < 9; ++r)
    Aw[((size_t)r * 1024 + oc) * 2176 + c] = (_Float16)src[r];
}

__global__ void pack_w2_kernel(const float* __restrict__ tw2, _Float16* __restrict__ A2, int n) {
  int i = blockIdx.x * 256 + threadIdx.x;
  if (i < n) A2[i] = (_Float16)tw2[i];
}

// ---------------- classifier 1x1 conv (no bias), fp32 for accuracy ----------------
// block per (b,y); wave-uniform weight loads; coalesced x5 reads across xi
__global__ __launch_bounds__(256)
void cls_logits_kernel(const float* __restrict__ x5, const float* __restrict__ clsw,
                       float* __restrict__ logits) {
  int b = blockIdx.x >> 5, y = blockIdx.x & 31;
  int xi = threadIdx.x & 31, g = threadIdx.x >> 5;
  if (g >= 7) return;  // 7 groups x 3 output channels = 21
  float a0 = 0.f, a1 = 0.f, a2 = 0.f;
  const float* xb = x5 + ((size_t)b * 2048) * 1024 + y * 32 + xi;
  const float* w0 = clsw + (size_t)(g * 3 + 0) * 2048;
  const float* w1 = clsw + (size_t)(g * 3 + 1) * 2048;
  const float* w2 = clsw + (size_t)(g * 3 + 2) * 2048;
  for (int c = 0; c < 2048; ++c) {
    float xv = xb[(size_t)c * 1024];
    a0 += xv * w0[c]; a1 += xv * w1[c]; a2 += xv * w2[c];
  }
  size_t ob = (((size_t)b * 21 + g * 3) * 32 + y) * 32 + xi;
  logits[ob] = a0; logits[ob + 1024] = a1; logits[ob + 2048] = a2;
}

// ---------------- global average pool: block per (b,c) ----------------
__global__ __launch_bounds__(256)
void gap_kernel(const float* __restrict__ x5, float* __restrict__ gap) {
  __shared__ float s[256];
  const size_t base = (size_t)blockIdx.x * 1024;
  float v = x5[base + threadIdx.x] + x5[base + threadIdx.x + 256] +
            x5[base + threadIdx.x + 512] + x5[base + threadIdx.x + 768];
  s[threadIdx.x] = v;
  __syncthreads();
  for (int st = 128; st > 0; st >>= 1) {
    if ((int)threadIdx.x < st) s[threadIdx.x] += s[threadIdx.x + st];
    __syncthreads();
  }
  if (threadIdx.x == 0) gap[blockIdx.x] = s[0] * (1.f / 1024.f);
}

__global__ void logits_min_kernel(const float* __restrict__ gap,
                                  const float* __restrict__ clsw,
                                  float* __restrict__ outp) {
  int t = blockIdx.x * 256 + threadIdx.x;
  if (t >= 168) return;
  int b = t / 21, oc = t % 21;
  float acc = 0.f;
  const float* g = gap + (size_t)b * 2048;
  const float* w = clsw + (size_t)oc * 2048;
  for (int c = 0; c < 2048; ++c) acc += g[c] * w[c];
  outp[t] = acc;
}

// ---------------- d3: 1x1 conv 256->18 (+relu), fp32 out NHWC ----------------
__global__ void d3_kernel(const _Float16* __restrict__ d2, const float* __restrict__ w3,
                          const float* __restrict__ b3, float* __restrict__ d3o) {
  int idx = blockIdx.x * 256 + threadIdx.x;
  if (idx >= 8192 * 18) return;
  int k = idx % 18, n = idx / 18;
  float acc = b3[k];
  const _Float16* row = d2 + (size_t)n * 256;
  const float* wr = w3 + (size_t)k * 256;
  for (int c = 0; c < 256; ++c) acc += (float)row[c] * wr[c];
  d3o[(size_t)n * 18 + k] = acc > 0.f ? acc : 0.f;
}

// ---------------- fused softmax + get_noliner + upfeat + concat ----------------
// out[b,ch,y,x]: ch=0 from logits[0]*softmax_bg, ch=1..20 from logits[1..20]*softmax_fg,
// gathered over the 9 in-bounds neighbors (OOB terms are zero in the reference).
__global__ void final_gather_kernel(const float* __restrict__ d3,
                                    const float* __restrict__ logits,
                                    float* __restrict__ out) {
  int n = blockIdx.x * 256 + threadIdx.x;
  if (n >= 8192) return;
  int b = n >> 10, y = (n >> 5) & 31, x = n & 31;
  float bg = 0.f, fg[20];
  #pragma unroll
  for (int c = 0; c < 20; ++c) fg[c] = 0.f;
  for (int i = 0; i < 3; ++i) {
    int yn = y + i - 1;
    if ((unsigned)yn >= 32u) continue;
    for (int j = 0; j < 3; ++j) {
      int xn = x + j - 1;
      if ((unsigned)xn >= 32u) continue;
      const float* dr = d3 + ((size_t)(b * 1024 + yn * 32 + xn)) * 18;
      int k = i * 3 + j;
      float m0 = dr[0];
      for (int q = 1; q < 9; ++q) m0 = fmaxf(m0, dr[q]);
      float s0 = 0.f;
      for (int q = 0; q < 9; ++q) s0 += __expf(dr[q] - m0);
      float wbg = __expf(dr[k] - m0) / s0;
      float m1 = dr[9];
      for (int q = 10; q < 18; ++q) m1 = fmaxf(m1, dr[q]);
      float s1 = 0.f;
      for (int q = 9; q < 18; ++q) s1 += __expf(dr[q] - m1);
      float wfg = __expf(dr[9 + k] - m1) / s1;
      const float* lr = logits + (size_t)b * 21 * 1024 + yn * 32 + xn;
      bg += wbg * lr[0];
      #pragma unroll
      for (int c = 0; c < 20; ++c) fg[c] += wfg * lr[(size_t)(1 + c) * 1024];
    }
  }
  size_t ob = (size_t)b * 21 * 1024 + (size_t)y * 32 + x;
  out[ob] = bg;
  #pragma unroll
  for (int c = 0; c < 20; ++c) out[ob + (size_t)(1 + c) * 1024] = fg[c];
}

// ---------------- launcher ----------------
extern "C" void kernel_launch(void* const* d_in, const int* in_sizes, int n_in,
                              void* d_out, int out_size, void* d_ws, size_t ws_size,
                              hipStream_t stream) {
  (void)in_sizes; (void)n_in; (void)out_size; (void)ws_size;
  const float* x5    = (const float*)d_in[0];
  const float* probs = (const float*)d_in[1];
  const float* qw1 = (const float*)d_in[2];  const float* qb1 = (const float*)d_in[3];
  const float* qw2 = (const float*)d_in[4];  const float* qb2 = (const float*)d_in[5];
  const float* qw3 = (const float*)d_in[6];  const float* qb3 = (const float*)d_in[7];
  const float* qw4 = (const float*)d_in[8];  const float* qb4 = (const float*)d_in[9];
  const float* tw1 = (const float*)d_in[10]; const float* tb1 = (const float*)d_in[11];
  const float* tw2 = (const float*)d_in[12]; const float* tb2 = (const float*)d_in[13];
  const float* tw3 = (const float*)d_in[14]; const float* tb3 = (const float*)d_in[15];
  const float* clsw = (const float*)d_in[16];
  float* out = (float*)d_out;

  char* ws = (char*)d_ws;
  float*    q1   = (float*)(ws + OFF_Q1);
  float*    q2   = (float*)(ws + OFF_Q2);
  float*    q3   = (float*)(ws + OFF_Q3);
  float*    q4   = (float*)(ws + OFF_Q4);
  _Float16* feat = (_Float16*)(ws + OFF_FEAT);
  _Float16* Aw1  = (_Float16*)(ws + OFF_AW1);
  _Float16* A2   = (_Float16*)(ws + OFF_A2);
  _Float16* d1   = (_Float16*)(ws + OFF_D1);
  _Float16* d2   = (_Float16*)(ws + OFF_D2);
  float*    d3   = (float*)(ws + OFF_D3);
  float*    lgt  = (float*)(ws + OFF_LOG);
  float*    gap  = (float*)(ws + OFF_GAP);

  // weight packing (f16, GEMM-friendly layouts)
  pack_w1_kernel<<<8704, 256, 0, stream>>>(tw1, Aw1);
  pack_w2_kernel<<<1024, 256, 0, stream>>>(tw2, A2, 256 * 1024);

  // classifier branch (independent)
  cls_logits_kernel<<<256, 256, 0, stream>>>(x5, clsw, lgt);
  gap_kernel<<<16384, 256, 0, stream>>>(x5, gap);
  logits_min_kernel<<<1, 256, 0, stream>>>(gap, clsw, out + 172032);

  // query feature chain q1..q4 (naive fp32; ~5 GFLOP total)
  conv_naive_relu_kernel<<<16384, 256, 0, stream>>>(probs, qw1, qb1, q1, 8, 9, 512, 32, 128, 4, 4, 1);
  conv_naive_relu_kernel<<<2048, 256, 0, stream>>>(q1, qw2, qb2, q2, 8, 32, 128, 64, 32, 4, 4, 1);
  conv_naive_relu_kernel<<<4096, 256, 0, stream>>>(q2, qw3, qb3, q3, 8, 64, 32, 128, 32, 3, 1, 1);
  conv_naive_relu_kernel<<<4096, 256, 0, stream>>>(q3, qw4, qb4, q4, 8, 128, 32, 128, 32, 3, 1, 1);

  // build padded NHWC f16 feature map: zero (for the pad ring) then transpose-fill
  zero_h_kernel<<<9826, 256, 0, stream>>>(feat, 20123648 / 8);
  pack_feat_kernel<<<17408, 256, 0, stream>>>(x5, q4, feat);

  // the big one: 3x3 conv 2176->1024 on v_wmma_f32_16x16x32_f16 (328 GFLOP)
  conv3x3_wmma_kernel<<<1024, 256, 0, stream>>>(Aw1, feat, tb1, d1);

  // 1x1 conv 1024->256 on WMMA
  gemm_nk_wmma_kernel<<<256, 256, 0, stream>>>(A2, d1, tb2, d2, 256, 1024, 2);

  // 1x1 conv 256->18 (+relu)
  d3_kernel<<<576, 256, 0, stream>>>(d2, tw3, tb3, d3);

  // fused softmax / get_noliner / upfeat / concat
  final_gather_kernel<<<32, 256, 0, stream>>>(d3, lgt, out);
}
